// SWEGNN_61624190763508
// MI455X (gfx1250) — matrix operand order repaired and checked by reading, not compiled, not run on hardware
//
#include <hip/hip_runtime.h>
#include <math.h>

// ---------------- problem constants (fixed by reference) ----------------
#define N_NODES   100000
#define N_EDGES   1600000
#define STATIC_F  16
#define DYN_F     32
#define EDGE_F    8
#define IN_F      104           // padded to 128 for WMMA K
#define IN_F_PAD  128
#define HID_F     64

// edge kernel tiling
#define WPB       4             // waves per block (wave32)
#define TPB       (WPB * 32)    // 128 threads
#define EDGES_PER_BLOCK (WPB * 16)   // 64
#define A_STRIDE  136           // halves, padded row stride of A tile
#define H_STRIDE  72            // halves, padded row stride of hidden tile

// workspace layout (bytes)
#define WS_W1F    0                         // 16 frags * 512 halves * 2B = 16384
#define WS_W2F    16384                     // 4 frags  * 512 halves * 2B =  4096
#define WS_MASK   20480                     // N_NODES * 4                = 400000
#define WS_SCAT   420480                    // N_NODES * 32 * 4           = 12.8MB

typedef __attribute__((ext_vector_type(16))) _Float16 v16h;
typedef __attribute__((ext_vector_type(8)))  float    v8f;

union HalfPair { unsigned int u; _Float16 h[2]; };

// ---- A-fragment (16x32 f16, MxK) loader from an LDS tile ----
// ISA 7.12.2 16-bit A layout: lane L holds row m=L%16; for half slot h
// (VGPR v=h/2, pos=h%2, group g=L/16):  K = (v<4 ? 2v : 16+2(v-4)) + pos + 8g
__device__ inline v16h load_a_frag(const _Float16* tile, int ktile, int lane, int stride) {
    const int m = lane & 15, g = lane >> 4;
    const unsigned int* rowp = (const unsigned int*)(tile + m * stride);
    v16h a;
#pragma unroll
    for (int v = 0; v < 8; ++v) {
        int kpair = (v < 4 ? 2 * v : 16 + 2 * (v - 4)) + 8 * g;   // even
        HalfPair hp; hp.u = rowp[(ktile * 32 + kpair) >> 1];
        a[2 * v]     = hp.h[0];
        a[2 * v + 1] = hp.h[1];
    }
    return a;
}

// ---------------- kernel 0: weight -> f16 B-fragment prep ----------------
// B (32x16 f16, KxN): lane L holds column n=L%16; half h holds K = (h<8 ? h : 16+(h-8)) + 8*(L/16).
// Fragment storage in ws: [frag][lane][16 halves] so a lane's v16h is contiguous.
__global__ void swegnn_prep(const float* __restrict__ W1, const float* __restrict__ W2,
                            _Float16* __restrict__ w1f, _Float16* __restrict__ w2f) {
    int t = blockIdx.x * blockDim.x + threadIdx.x;
    if (t < 16 * 512) {                 // W1: 4 ktiles x 4 ntiles
        int f = t >> 9, r = t & 511, lane = r >> 4, h = r & 15;
        int nt = f & 3, kt = f >> 2;
        int n = nt * 16 + (lane & 15);
        int k = kt * 32 + (h < 8 ? h : 16 + (h - 8)) + (lane >> 4) * 8;
        float v = (k < IN_F) ? W1[k * HID_F + n] : 0.0f;
        w1f[t] = (_Float16)v;
    } else if (t < 16 * 512 + 4 * 512) { // W2: 2 ktiles x 2 ntiles
        int t2 = t - 16 * 512;
        int f = t2 >> 9, r = t2 & 511, lane = r >> 4, h = r & 15;
        int nt = f & 1, kt = f >> 1;
        int n = nt * 16 + (lane & 15);
        int k = kt * 32 + (h < 8 ? h : 16 + (h - 8)) + (lane >> 4) * 8;
        w2f[t2] = (_Float16)W2[k * DYN_F + n];
    }
}

// ---------------- out = (base?) + sin @ Fk   ([N,32] = [N,32]@[32,32]) ----------------
__global__ void swegnn_update(const float* __restrict__ sin_, const float* __restrict__ Fk,
                              const float* __restrict__ base, float* __restrict__ outp) {
    __shared__ float shF[32 * 32];
    for (int i = threadIdx.x; i < 32 * 32; i += blockDim.x) shF[i] = Fk[i];
    __syncthreads();
    int m = threadIdx.x >> 5, j = threadIdx.x & 31;
    int node = blockIdx.x * 8 + m;
    if (node < N_NODES) {
        float acc = base ? base[node * 32 + j] : 0.0f;
        const float* srow = sin_ + node * 32;
#pragma unroll
        for (int i = 0; i < 32; ++i) acc = fmaf(srow[i], shF[i * 32 + j], acc);
        outp[node * 32 + j] = acc;
    }
}

// ---------------- mask[n] = (sum_j out[n][j]) != 0 ----------------
__global__ void swegnn_mask(const float* __restrict__ outp, int* __restrict__ maskbuf) {
    int n = blockIdx.x * blockDim.x + threadIdx.x;
    if (n < N_NODES) {
        float s = 0.0f;
        const float* r = outp + n * 32;
#pragma unroll
        for (int j = 0; j < 32; ++j) s += r[j];
        maskbuf[n] = (s != 0.0f) ? 1 : 0;
    }
}

__global__ void swegnn_zero(float* __restrict__ p, int n) {
    int i = blockIdx.x * blockDim.x + threadIdx.x;
    if (i < n) p[i] = 0.0f;
}

// ---------------- edge MLP kernel: 16 edges per wave, WMMA f16->f32 ----------------
__global__ void __launch_bounds__(TPB)
swegnn_edge(const float* __restrict__ xs, const int* __restrict__ ei,
            const float* __restrict__ ea, const float* __restrict__ b1,
            const float* __restrict__ b2, const float* __restrict__ outp,
            const _Float16* __restrict__ w1f, const _Float16* __restrict__ w2f,
            const int* __restrict__ maskbuf, float* __restrict__ scat) {
    __shared__ _Float16 shA[WPB][16][A_STRIDE];   // f16 A tile: [edge][K 0..127]
    __shared__ float    shOr[WPB][16][32];        // out[row] (f32, for diff)
    __shared__ float    shOc[WPB][16][32];        // out[col]
    __shared__ _Float16 shH[WPB][16][H_STRIDE];   // hidden activations (f16)
    __shared__ float    shW[WPB][16][32];         // GEMM2 result (f32)
    __shared__ int      shEm[WPB][2][16];         // per-edge mask halves

    const int tid  = threadIdx.x;
    const int w    = tid >> 5;          // wave in block
    const int lane = tid & 31;
    const int m    = lane & 15;         // edge within tile
    const int g    = lane >> 4;         // half-group
    const int ebase = blockIdx.x * EDGES_PER_BLOCK + w * 16;
    const int e     = ebase + m;
    const bool ok   = (e < N_EDGES);

    // ---- gather phase: build A row [xs_row | xs_col | out_row | out_col | ea | 0pad] ----
    if (ok) {
        if (g == 0) {
            int row = ei[e];
            const float* xr = xs + row * STATIC_F;
#pragma unroll
            for (int j = 0; j < STATIC_F; ++j) shA[w][m][j] = (_Float16)xr[j];
            const float* orw = outp + row * DYN_F;
#pragma unroll
            for (int j = 0; j < DYN_F; ++j) {
                float v = orw[j];
                shA[w][m][32 + j] = (_Float16)v;
                shOr[w][m][j] = v;
            }
            const float* eap = ea + e * EDGE_F;
#pragma unroll
            for (int j = 0; j < EDGE_F; ++j) shA[w][m][96 + j] = (_Float16)eap[j];
#pragma unroll
            for (int j = IN_F; j < IN_F_PAD; ++j) shA[w][m][j] = (_Float16)0.0f;
            shEm[w][0][m] = maskbuf[row];
        } else {
            int col = ei[N_EDGES + e];
            const float* xc = xs + col * STATIC_F;
#pragma unroll
            for (int j = 0; j < STATIC_F; ++j) shA[w][m][16 + j] = (_Float16)xc[j];
            const float* ocl = outp + col * DYN_F;
#pragma unroll
            for (int j = 0; j < DYN_F; ++j) {
                float v = ocl[j];
                shA[w][m][64 + j] = (_Float16)v;
                shOc[w][m][j] = v;
            }
            shEm[w][1][m] = maskbuf[col];
        }
    }
    __syncthreads();

    // ---- GEMM1: [16,128] x [128,64] -> 4 accum tiles ----
    const v16h* w1v = (const v16h*)w1f;
    v8f c[4];
#pragma unroll
    for (int nt = 0; nt < 4; ++nt) c[nt] = (v8f){0, 0, 0, 0, 0, 0, 0, 0};
#pragma unroll
    for (int kt = 0; kt < 4; ++kt) {
        v16h a = load_a_frag(&shA[w][0][0], kt, lane, A_STRIDE);
#pragma unroll
        for (int nt = 0; nt < 4; ++nt) {
            v16h b = w1v[(kt * 4 + nt) * 32 + lane];
            c[nt] = __builtin_amdgcn_wmma_f32_16x16x32_f16(
                false, a, false, b, (short)0, c[nt], false, false);
        }
    }
    // bias + ReLU -> f16 hidden tile. C/D layout: elem r -> (M=r+8g, N=lane%16)
#pragma unroll
    for (int nt = 0; nt < 4; ++nt) {
        int hid = nt * 16 + m;
        float bb = b1[hid];
#pragma unroll
        for (int r = 0; r < 8; ++r) {
            float v = c[nt][r] + bb;
            shH[w][r + 8 * g][hid] = (_Float16)(v > 0.0f ? v : 0.0f);
        }
    }
    __syncthreads();

    // ---- GEMM2: [16,64] x [64,32] -> 2 accum tiles ----
    const v16h* w2v = (const v16h*)w2f;
    v8f d[2];
#pragma unroll
    for (int nt = 0; nt < 2; ++nt) d[nt] = (v8f){0, 0, 0, 0, 0, 0, 0, 0};
#pragma unroll
    for (int kt = 0; kt < 2; ++kt) {
        v16h a2 = load_a_frag(&shH[w][0][0], kt, lane, H_STRIDE);
#pragma unroll
        for (int nt = 0; nt < 2; ++nt) {
            v16h b = w2v[(kt * 2 + nt) * 32 + lane];
            d[nt] = __builtin_amdgcn_wmma_f32_16x16x32_f16(
                false, a2, false, b, (short)0, d[nt], false, false);
        }
    }
#pragma unroll
    for (int nt = 0; nt < 2; ++nt) {
        int n = nt * 16 + m;
        float bb = b2[n];
#pragma unroll
        for (int r = 0; r < 8; ++r) shW[w][r + 8 * g][n] = d[nt][r] + bb;
    }
    __syncthreads();

    // ---- normalize + masked diff + atomic scatter ----
    if (ok) {
        float nrm2 = 0.0f;
#pragma unroll
        for (int j = 0; j < 32; ++j) { float v = shW[w][m][j]; nrm2 = fmaf(v, v, nrm2); }
        float nrm = sqrtf(nrm2);
        float scale = (nrm > 0.0f) ? (1.0f / nrm) : 0.0f;
        float em = (shEm[w][0][m] | shEm[w][1][m]) ? 1.0f : 0.0f;
        float f = scale * em;
        if (f != 0.0f) {
            int col = ei[N_EDGES + e];
            float* dst = scat + col * 32;
#pragma unroll
            for (int jj = 0; jj < 16; ++jj) {
                int j = g * 16 + jj;
                float s = (shOc[w][m][j] - shOr[w][m][j]) * shW[w][m][j] * f;
                atomicAdd(&dst[j], s);
            }
        }
    }
}

// ---------------- host-side launch ----------------
extern "C" void kernel_launch(void* const* d_in, const int* in_sizes, int n_in,
                              void* d_out, int out_size, void* d_ws, size_t ws_size,
                              hipStream_t stream) {
    const float* xs = (const float*)d_in[0];
    const float* xt = (const float*)d_in[1];
    const int*   ei = (const int*)d_in[2];
    const float* ea = (const float*)d_in[3];
    const float* W1 = (const float*)d_in[4];
    const float* b1 = (const float*)d_in[5];
    const float* W2 = (const float*)d_in[6];
    const float* b2 = (const float*)d_in[7];
    const float* F  = (const float*)d_in[8];
    float* outp = (float*)d_out;

    char* ws = (char*)d_ws;
    _Float16* w1f     = (_Float16*)(ws + WS_W1F);
    _Float16* w2f     = (_Float16*)(ws + WS_W2F);
    int*      maskbuf = (int*)(ws + WS_MASK);
    float*    scat    = (float*)(ws + WS_SCAT);

    // 0) weight fragments (f16, WMMA B layout)
    swegnn_prep<<<40, 256, 0, stream>>>(W1, W2, w1f, w2f);

    // 1) out = x_t @ F[0]
    swegnn_update<<<(N_NODES + 7) / 8, 256, 0, stream>>>(xt, F, nullptr, outp);

    const int edge_blocks = (N_EDGES + EDGES_PER_BLOCK - 1) / EDGES_PER_BLOCK;
    for (int k = 0; k < 2; ++k) {
        swegnn_mask<<<(N_NODES + 255) / 256, 256, 0, stream>>>(outp, maskbuf);
        swegnn_zero<<<(N_NODES * 32 + 255) / 256, 256, 0, stream>>>(scat, N_NODES * 32);
        swegnn_edge<<<edge_blocks, TPB, 0, stream>>>(xs, ei, ea, b1, b2, outp,
                                                     w1f, w2f, maskbuf, scat);
        swegnn_update<<<(N_NODES + 7) / 8, 256, 0, stream>>>(
            scat, F + (size_t)(k + 1) * 32 * 32, outp, outp);
    }
}
